// MoEVM_62380105007239
// MI455X (gfx1250) — compile-verified, matrix operand order checked by reading, not run on hardware
//
#include <hip/hip_runtime.h>
#include <hip/hip_bf16.h>

typedef __attribute__((ext_vector_type(16))) _Float16 v16h;
typedef __attribute__((ext_vector_type(8)))  float    v8f;

#define WAVES_PER_BLOCK 4
#define ROWS_PER_WAVE   32
#define BLOCK_THREADS   (WAVES_PER_BLOCK * 32)
#define BLOCK_ROWS      (WAVES_PER_BLOCK * ROWS_PER_WAVE)
#define SCALE_T         100.0f

struct WaveLds {
  float scr[32][48];     // stage1: [0..15]=HI pre, [16..31]=LO pre ; stage3: F0 | F1 | XOR
  float nib[32][4][16];  // nibble dists per row: 0=ah 1=al 2=bh 3=bl
};

// ---- WMMA helpers -----------------------------------------------------------

__device__ __forceinline__ v8f wmma_f16(v16h a, v16h b, v8f c) {
  return __builtin_amdgcn_wmma_f32_16x16x32_f16(false, a, false, b, (short)0, c,
                                                false, false);
}

// 16-bit A fragment (16x32): lane m=lane&15 is row M; khalf=lane>>4 selects
// K = 32c + 8*khalf + (e<8 ? e : e+8)  -> two contiguous runs of 8 floats.
__device__ __forceinline__ v16h load_a_frag(const float* __restrict__ rp, int c, int kh) {
  const float* p = rp + 32 * c + 8 * kh;
  v16h f;
#pragma unroll
  for (int e = 0; e < 8; ++e) {
    f[e]     = (_Float16)p[e];
    f[e + 8] = (_Float16)p[e + 16];
  }
  return f;
}

// B fragment (32x16) generated from index predicates (0/1 tables, exact in f16).
// TBL: 0 = high-nibble select, 1 = low-nibble select,
//      2 = (hi+lo)==n (full-sum cols 0..15), 3 = (hi+lo)==n+16 (cols 16..30),
//      4 = (hi^lo)==n (XOR table).
template <int TBL>
__device__ __forceinline__ v16h make_b_frag(int c, int n, int kh) {
  v16h f;
#pragma unroll
  for (int e = 0; e < 16; ++e) {
    int k  = 32 * c + 8 * kh + (e < 8 ? e : e + 8);
    int hi = k >> 4, lo = k & 15;
    bool on;
    if      (TBL == 0) on = (hi == n);
    else if (TBL == 1) on = (lo == n);
    else if (TBL == 2) on = (hi + lo == n);
    else if (TBL == 3) on = (hi + lo == n + 16);
    else               on = ((hi ^ lo) == n);
    f[e] = on ? (_Float16)1.0f : (_Float16)0.0f;
  }
  return f;
}

// ---- VALU helpers -----------------------------------------------------------

__device__ __forceinline__ void softmax16(float v[16]) {
  float mx = v[0];
#pragma unroll
  for (int i = 1; i < 16; ++i) mx = fmaxf(mx, v[i]);
  float s = 0.f;
#pragma unroll
  for (int i = 0; i < 16; ++i) { v[i] = __expf(SCALE_T * (v[i] - mx)); s += v[i]; }
  float inv = 1.f / s;
#pragma unroll
  for (int i = 0; i < 16; ++i) v[i] *= inv;
}

__device__ __forceinline__ void softmax16_ld_st(const float* __restrict__ in,
                                                float* __restrict__ outp) {
  float v[16];
#pragma unroll
  for (int i = 0; i < 16; ++i) v[i] = in[i];
  softmax16(v);
#pragma unroll
  for (int i = 0; i < 16; ++i) outp[i] = v[i];
}

// softmax over 256 of (h[y>>4] + l[y&15]) is separable: outer product of the
// two 16-way softmax factors. 32 exps instead of 256, contiguous b128 stores.
__device__ __forceinline__ void write_byte(float* __restrict__ o,
                                           const float h[16], const float l[16]) {
  float mh = h[0], ml = l[0];
#pragma unroll
  for (int i = 1; i < 16; ++i) { mh = fmaxf(mh, h[i]); ml = fmaxf(ml, l[i]); }
  float eh[16], el[16];
  float sh = 0.f, sl = 0.f;
#pragma unroll
  for (int i = 0; i < 16; ++i) {
    eh[i] = __expf(SCALE_T * (h[i] - mh)); sh += eh[i];
    el[i] = __expf(SCALE_T * (l[i] - ml)); sl += el[i];
  }
  float inv = 1.f / (sh * sl);
#pragma unroll
  for (int i = 0; i < 16; ++i) {
    float hv = eh[i] * inv;
#pragma unroll
    for (int q = 0; q < 4; ++q) {
      float4 v;
      v.x = hv * el[4 * q + 0];
      v.y = hv * el[4 * q + 1];
      v.z = hv * el[4 * q + 2];
      v.w = hv * el[4 * q + 3];
      reinterpret_cast<float4*>(o)[i * 4 + q] = v;
    }
  }
}

// ---- Kernel -----------------------------------------------------------------

__global__ __launch_bounds__(BLOCK_THREADS, 1)
void moealu_kernel(const float* __restrict__ Ain, const float* __restrict__ Bin,
                   float* __restrict__ out, int Bn) {
  __shared__ WaveLds lds[WAVES_PER_BLOCK];
  const int tid  = threadIdx.x;
  const int wave = tid >> 5;
  const int lane = tid & 31;
  const int m    = lane & 15;   // fragment row / column index
  const int kh   = lane >> 4;   // K-half within fragments
  WaveLds& W = lds[wave];

  const int rowBase = blockIdx.x * BLOCK_ROWS + wave * ROWS_PER_WAVE;
  const int myrow   = rowBase + lane;   // row owned by this lane for scalar stages

  float c0 = 1.f, c1 = 0.f;             // carry distribution [P(0), P(1)]

  for (int byte = 0; byte < 4; ++byte) {
    if (byte < 3) {
      __builtin_prefetch(Ain + (((size_t)myrow * 4 + byte + 1) << 8), 0, 1);
      __builtin_prefetch(Bin + (((size_t)myrow * 4 + byte + 1) << 8), 0, 1);
    }

    // ---- stage 1: to_nibbles via WMMA ([16x256] x [256x16] HI and LO) ------
#pragma unroll
    for (int op = 0; op < 2; ++op) {
      const float* src = op ? Bin : Ain;
#pragma unroll
      for (int t = 0; t < 2; ++t) {
        const float* rp =
            src + (((size_t)(rowBase + 16 * t + m) * 4 + byte) << 8);
        v8f accH = {}; v8f accL = {};
#pragma unroll
        for (int c = 0; c < 8; ++c) {
          v16h af = load_a_frag(rp, c, kh);
          accH = wmma_f16(af, make_b_frag<0>(c, m, kh), accH);
          accL = wmma_f16(af, make_b_frag<1>(c, m, kh), accL);
        }
#pragma unroll
        for (int r = 0; r < 8; ++r) {
          int row = 16 * t + r + 8 * kh;      // C layout: M = r + 8*(lane>=16)
          W.scr[row][m]      = accH[r];
          W.scr[row][16 + m] = accL[r];
        }
      }
      __syncthreads();
      softmax16_ld_st(&W.scr[lane][0],  &W.nib[lane][2 * op + 0][0]); // high
      softmax16_ld_st(&W.scr[lane][16], &W.nib[lane][2 * op + 1][0]); // low
      __syncthreads();
    }

    // ---- stage 3: outer-product GEMMs vs full-sum (F0,F1) and XOR tables ---
    float s_lo[16], s_hi[16], x_lo[16], x_hi[16];
#pragma unroll
    for (int pair = 0; pair < 2; ++pair) {
      const int xs = pair ? 0 : 1;   // pair0: (al,bl)  pair1: (ah,bh)
      const int ys = pair ? 2 : 3;
#pragma unroll
      for (int t = 0; t < 2; ++t) {
        const int frow = 16 * t + m;
        float x16[16], y8[8];
#pragma unroll
        for (int i = 0; i < 16; ++i) x16[i] = W.nib[frow][xs][i];
#pragma unroll
        for (int e = 0; e < 8; ++e)  y8[e]  = W.nib[frow][ys][8 * kh + e];
        v8f aF0 = {}; v8f aF1 = {}; v8f aX = {};
#pragma unroll
        for (int c = 0; c < 8; ++c) {
          // outer-product A fragment: K = i*16+j, i = 2c+(e>=8), j = 8*kh+(e&7)
          v16h of;
          float x0 = x16[2 * c], x1 = x16[2 * c + 1];
#pragma unroll
          for (int e = 0; e < 8; ++e) {
            of[e]     = (_Float16)(x0 * y8[e]);
            of[e + 8] = (_Float16)(x1 * y8[e]);
          }
          aF0 = wmma_f16(of, make_b_frag<2>(c, m, kh), aF0);
          aF1 = wmma_f16(of, make_b_frag<3>(c, m, kh), aF1);
          aX  = wmma_f16(of, make_b_frag<4>(c, m, kh), aX);
        }
#pragma unroll
        for (int r = 0; r < 8; ++r) {
          int row = 16 * t + r + 8 * kh;
          W.scr[row][m]      = aF0[r];
          W.scr[row][16 + m] = aF1[r];
          W.scr[row][32 + m] = aX[r];
        }
      }
      __syncthreads();

      // per-lane: carry blend + softmaxes on its own row
      float f[31];
#pragma unroll
      for (int n = 0; n < 16; ++n) f[n] = W.scr[lane][n];
#pragma unroll
      for (int n = 0; n < 15; ++n) f[16 + n] = W.scr[lane][16 + n];
      float g0[16];
#pragma unroll
      for (int n = 0; n < 15; ++n) g0[n] = f[n] + f[n + 16];
      g0[15] = f[15];

      float sp[16];
#pragma unroll
      for (int l = 0; l < 16; ++l)
        sp[l] = c0 * g0[l] + c1 * g0[(l + 15) & 15];

      float sge = 0.f, slt = 0.f;
#pragma unroll
      for (int n = 16; n < 31; ++n) sge += f[n];
#pragma unroll
      for (int n = 0; n < 16; ++n) slt += f[n];
      float p1 = c0 * sge + c1 * (sge + f[15]);
      float p0 = c0 * slt + c1 * (slt - f[15]);
      float mx = fmaxf(p0, p1);
      float e0 = __expf(SCALE_T * (p0 - mx));
      float e1 = __expf(SCALE_T * (p1 - mx));
      float iv = 1.f / (e0 + e1);
      c0 = e0 * iv; c1 = e1 * iv;

      softmax16(sp);
      float xp[16];
#pragma unroll
      for (int n = 0; n < 16; ++n) xp[n] = W.scr[lane][32 + n];
      softmax16(xp);

      if (pair == 0) {
#pragma unroll
        for (int n = 0; n < 16; ++n) { s_lo[n] = sp[n]; x_lo[n] = xp[n]; }
      } else {
#pragma unroll
        for (int n = 0; n < 16; ++n) { s_hi[n] = sp[n]; x_hi[n] = xp[n]; }
      }
      __syncthreads();
    }

    // ---- stage 4: from_nibbles (separable softmax-256) + store -------------
    write_byte(out + (((size_t)myrow * 4 + byte) << 8),        s_hi, s_lo);
    write_byte(out + (((size_t)(Bn + myrow) * 4 + byte) << 8), x_hi, x_lo);
  }
}

extern "C" void kernel_launch(void* const* d_in, const int* in_sizes, int n_in,
                              void* d_out, int out_size, void* d_ws, size_t ws_size,
                              hipStream_t stream) {
  (void)n_in; (void)out_size; (void)d_ws; (void)ws_size;
  const float* a = (const float*)d_in[0];
  const float* b = (const float*)d_in[1];
  float* out = (float*)d_out;
  int Bn = in_sizes[0] / 1024;          // [B,4,256] -> B
  int grid = Bn / BLOCK_ROWS;
  if (grid < 1) grid = 1;
  moealu_kernel<<<grid, BLOCK_THREADS, 0, stream>>>(a, b, out, Bn);
}
